// DeeperGCN_8246337208545
// MI455X (gfx1250) — matrix operand order repaired and checked by reading, not compiled, not run on hardware
//
#include <hip/hip_runtime.h>

typedef __attribute__((ext_vector_type(16))) _Float16 v16h;
typedef __attribute__((ext_vector_type(8)))  _Float16 h8;
typedef __attribute__((ext_vector_type(8)))  float    v8f;

#define DH 64

// ---------------------------------------------------------------------------
// Fused WMMA GEMM: OUT[r, 0..NC) = A(r,:) @ W[:,0..NC) + bias (+ residual)
// MODE 0: A = x                         (encoder, KDIM=128)
// MODE 1: A = hin + num/(den+1e-16)     (GENConv update, KDIM=64)
// MODE 2: A = hin                       (predictor, KDIM=64, NC=40)
// Block = 256 threads = 8 waves; each wave owns a 16-row strip x NC cols.
// ---------------------------------------------------------------------------
template<int MODE, int KDIM, int NC, bool RES>
__global__ __launch_bounds__(256) void gemm_wmma(
    const float* __restrict__ Asrc, const float* __restrict__ num,
    const float* __restrict__ den,  const float* __restrict__ W,
    const float* __restrict__ bias, const float* __restrict__ hres,
    float* __restrict__ out, int nrows)
{
    constexpr int NT  = (NC + 15) / 16;   // 16-col output tiles
    constexpr int NCP = NT * 16;          // padded col count
    constexpr int LDT = 40;               // LDS row stride (halves), 16B aligned

    __shared__ __align__(16) _Float16 As[128 * LDT];
    __shared__ __align__(16) _Float16 Ws[NCP * LDT];

    const int tid  = threadIdx.x;
    const int lane = tid & 31, w = tid >> 5;
    const int r    = lane & 15, g = lane >> 4;
    const int brow = blockIdx.x * 128;

    v8f acc[NT] = {};

    for (int kc = 0; kc < KDIM; kc += 32) {
        // ---- stage A tile (128 x 32), f32 -> f16, row-major -------------
        for (int lin = tid; lin < 128 * 32; lin += 256) {
            int row = lin >> 5, col = lin & 31;
            int grow = brow + row;
            float v = 0.f;
            if (grow < nrows) {
                if (MODE == 0) {
                    v = Asrc[(size_t)grow * KDIM + kc + col];
                } else if (MODE == 1) {
                    int o = grow * DH + kc + col;
                    v = Asrc[o] + num[o] / (den[o] + 1e-16f);
                } else {
                    v = Asrc[(size_t)grow * DH + kc + col];
                }
            }
            As[row * LDT + col] = (_Float16)v;
        }
        // ---- stage W tile transposed: Ws[n][kk] = W[kc+kk][n] -----------
        for (int lin = tid; lin < 32 * NCP; lin += 256) {
            int kk = lin / NCP, n = lin % NCP;
            float v = (n < NC) ? W[(kc + kk) * NC + n] : 0.f;
            Ws[n * LDT + kk] = (_Float16)v;
        }
        __syncthreads();

        // ---- pack A per ISA 16-bit 16x32 layout: two contiguous 8-half runs
        v16h a;
        {
            const h8 a0 = *(const h8*)&As[(w * 16 + r) * LDT + 8 * g];
            const h8 a1 = *(const h8*)&As[(w * 16 + r) * LDT + 16 + 8 * g];
#pragma unroll
            for (int i = 0; i < 8; ++i) { a[i] = a0[i]; a[i + 8] = a1[i]; }
        }
#pragma unroll
        for (int t = 0; t < NT; ++t) {
            // B 32x16: lane holds column n=lane%16, k = 8g..8g+7, 16+8g..16+8g+7 (Wt tile)
            const h8 b0 = *(const h8*)&Ws[(t * 16 + r) * LDT + 8 * g];
            const h8 b1 = *(const h8*)&Ws[(t * 16 + r) * LDT + 16 + 8 * g];
            v16h b;
#pragma unroll
            for (int i = 0; i < 8; ++i) { b[i] = b0[i]; b[i + 8] = b1[i]; }
            acc[t] = __builtin_amdgcn_wmma_f32_16x16x32_f16(
                false, a, false, b, (short)0, acc[t], false, false);
        }
        __syncthreads();
    }

    // ---- store: D layout M = i + 8g, N = lane%16 ------------------------
    const int trow = brow + w * 16;           // N % 16 == 0 -> uniform guard
    if (trow < nrows) {
#pragma unroll
        for (int t = 0; t < NT; ++t) {
            int col = t * 16 + r;
            if (col < NC) {
                float bv = bias[col];
#pragma unroll
                for (int i = 0; i < 8; ++i) {
                    int row = trow + i + 8 * g;
                    float v = acc[t][i] + bv;
                    if (RES) v += hres[(size_t)row * NC + col];
                    out[(size_t)row * NC + col] = v;
                }
            }
        }
    }
}

// ---------------------------------------------------------------------------
// Single-pass edge aggregation.  Softmax is invariant to a constant shift, so
// instead of a segment-max pass we use a fixed shift C=4 (messages are
// relu(standardized activations)+1e-7, i.e. O(1..6); exponentials stay well
// inside f32 range).  agg = num/(den+1e-16) with num = sum(e*msg), den = sum(e).
// One wave per edge: 32 lanes x 2 contiguous features, L2-resident atomics.
// ---------------------------------------------------------------------------
__global__ __launch_bounds__(256) void init_agg(
    float* __restrict__ num, float* __restrict__ den, int total)
{
    int i = blockIdx.x * 256 + threadIdx.x;
    if (i < total) { num[i] = 0.f; den[i] = 0.f; }
}

__global__ __launch_bounds__(256) void edge_agg_k(
    const int* __restrict__ src, const int* __restrict__ dst,
    const float* __restrict__ hin, float* __restrict__ num,
    float* __restrict__ den, int E)
{
    int idx = blockIdx.x * 256 + threadIdx.x;   // 32 threads per edge
    int e = idx >> 5;
    if (e >= E) return;
    int fp = (idx & 31) << 1;
    int s = src[e], d = dst[e];
    const float2 hv = *(const float2*)(hin + (size_t)s * DH + fp);
    float msg0 = fmaxf(hv.x, 0.f) + 1e-7f;
    float msg1 = fmaxf(hv.y, 0.f) + 1e-7f;
    float e0 = __expf(msg0 - 4.0f);
    float e1 = __expf(msg1 - 4.0f);
    size_t o = (size_t)d * DH + fp;
    atomicAdd(den + o,     e0);
    atomicAdd(den + o + 1, e1);
    atomicAdd(num + o,     e0 * msg0);
    atomicAdd(num + o + 1, e1 * msg1);
}

// ---------------------------------------------------------------------------
// BatchNorm (training stats) + ReLU
// ---------------------------------------------------------------------------
__global__ void zero_stats_k(float* __restrict__ stats)
{
    stats[threadIdx.x] = 0.f;   // <<<1,128>>>
}

__global__ __launch_bounds__(256) void bn_stats_k(
    const float* __restrict__ h, float* __restrict__ stats, int nrows)
{
    __shared__ float ss[256], sq[256];
    int tid = threadIdx.x;
    int f = tid & 63, gr = tid >> 6;
    float s = 0.f, q = 0.f;
    int r0 = blockIdx.x * 1024 + gr;
    for (int j = 0; j < 256; ++j) {
        int r = r0 + j * 4;
        if (r < nrows) {
            float v = h[(size_t)r * DH + f];
            s += v; q += v * v;
        }
    }
    ss[tid] = s; sq[tid] = q;
    __syncthreads();
    if (tid < 64) {
        s = ss[tid] + ss[tid + 64] + ss[tid + 128] + ss[tid + 192];
        q = sq[tid] + sq[tid + 64] + sq[tid + 128] + sq[tid + 192];
        atomicAdd(stats + tid,      s);
        atomicAdd(stats + 64 + tid, q);
    }
}

__global__ __launch_bounds__(256) void bn_relu_k(
    const float* __restrict__ h, const float* __restrict__ stats,
    const float* __restrict__ gamma, const float* __restrict__ beta,
    float* __restrict__ h2, int nrows)
{
    int i = blockIdx.x * 256 + threadIdx.x;
    if (i >= nrows * DH) return;
    int f = i & 63;
    float inv_n = 1.0f / nrows;
    float mu  = stats[f] * inv_n;
    float var = stats[64 + f] * inv_n - mu * mu;
    float v = gamma[f] * (h[i] - mu) * rsqrtf(var + 1e-5f) + beta[f];
    h2[i] = fmaxf(v, 0.f);
}

// ---------------------------------------------------------------------------
// log_softmax over 40 logits, in place
// ---------------------------------------------------------------------------
__global__ __launch_bounds__(256) void lsm_k(float* __restrict__ out, int nrows)
{
    int r = blockIdx.x * 256 + threadIdx.x;
    if (r >= nrows) return;
    float* p = out + (size_t)r * 40;
    float mx = -1e30f;
    for (int i = 0; i < 40; ++i) mx = fmaxf(mx, p[i]);
    float s = 0.f;
    for (int i = 0; i < 40; ++i) s += __expf(p[i] - mx);
    float lse = mx + __logf(s);
    for (int i = 0; i < 40; ++i) p[i] -= lse;
}

// ---------------------------------------------------------------------------
extern "C" void kernel_launch(void* const* d_in, const int* in_sizes, int n_in,
                              void* d_out, int out_size, void* d_ws, size_t ws_size,
                              hipStream_t stream)
{
    const float* x      = (const float*)d_in[0];
    const int*   ei     = (const int*)d_in[1];
    const float* W_enc  = (const float*)d_in[2];
    const float* b_enc  = (const float*)d_in[3];
    const float* Wg     = (const float*)d_in[4];
    const float* bg     = (const float*)d_in[5];
    const float* gamma  = (const float*)d_in[6];
    const float* beta   = (const float*)d_in[7];
    const float* W_pred = (const float*)d_in[8];
    const float* b_pred = (const float*)d_in[9];

    const int N  = in_sizes[0] / 128;
    const int E  = in_sizes[1] / 2;
    const int L  = in_sizes[5] / DH;      // bg is (L, 64)
    const int n64 = N * DH;

    const int* srcI = ei;
    const int* dstI = ei + E;

    float*    base   = (float*)d_ws;
    float*    h      = base;
    float*    h2     = base + (size_t)n64;
    float*    num    = base + 2 * (size_t)n64;
    float*    den    = base + 3 * (size_t)n64;
    float*    stats  = base + 4 * (size_t)n64;
    float*    logits = (float*)d_out;

    const int gemmGrid = (N + 127) / 128;
    const int elemGrid = (n64 + 255) / 256;
    const int edgeGrid = (E * 32 + 255) / 256;
    const int bnGrid   = (N + 1023) / 1024;
    dim3 B(256);

    // node encoder: h = x @ W_enc + b_enc
    gemm_wmma<0, 128, 64, false><<<gemmGrid, B, 0, stream>>>(
        x, nullptr, nullptr, W_enc, b_enc, nullptr, h, N);

    for (int l = 0; l < L; ++l) {
        const float* hin = h;
        if (l > 0) {
            zero_stats_k<<<1, 128, 0, stream>>>(stats);
            bn_stats_k<<<bnGrid, B, 0, stream>>>(h, stats, N);
            bn_relu_k<<<elemGrid, B, 0, stream>>>(
                h, stats, gamma + (l - 1) * DH, beta + (l - 1) * DH, h2, N);
            hin = h2;
        }
        init_agg<<<elemGrid, B, 0, stream>>>(num, den, n64);
        edge_agg_k<<<edgeGrid, B, 0, stream>>>(srcI, dstI, hin, num, den, E);
        if (l == 0) {
            gemm_wmma<1, 64, 64, false><<<gemmGrid, B, 0, stream>>>(
                hin, num, den, Wg + (size_t)l * DH * DH, bg + l * DH,
                nullptr, h, N);
        } else {
            gemm_wmma<1, 64, 64, true><<<gemmGrid, B, 0, stream>>>(
                hin, num, den, Wg + (size_t)l * DH * DH, bg + l * DH,
                h, h, N);
        }
    }

    // final BN -> ReLU -> predictor -> log_softmax
    zero_stats_k<<<1, 128, 0, stream>>>(stats);
    bn_stats_k<<<bnGrid, B, 0, stream>>>(h, stats, N);
    bn_relu_k<<<elemGrid, B, 0, stream>>>(
        h, stats, gamma + (L - 1) * DH, beta + (L - 1) * DH, h2, N);

    gemm_wmma<2, 64, 40, false><<<gemmGrid, B, 0, stream>>>(
        h2, nullptr, nullptr, W_pred, b_pred, nullptr, logits, N);

    lsm_k<<<(N + 255) / 256, B, 0, stream>>>(logits, N);
}
